// Attention_77730318123148
// MI455X (gfx1250) — matrix operand order repaired and checked by reading, not compiled, not run on hardware
//
#include <hip/hip_runtime.h>

// ---------------------------------------------------------------------------
// Types
// ---------------------------------------------------------------------------
typedef __attribute__((ext_vector_type(16))) _Float16 v16h;
typedef __attribute__((ext_vector_type(8)))  _Float16 v8h;
typedef __attribute__((ext_vector_type(4)))  _Float16 v4h;
typedef __attribute__((ext_vector_type(8)))  float    v8f;
typedef __attribute__((ext_vector_type(4)))  float    v4f;

// D = A(16x32 f16) * B(32x16 f16) + C(16x16 f32)
__device__ __forceinline__ v8f wmma_f16(v16h a, v16h b, v8f c) {
    return __builtin_amdgcn_wmma_f32_16x16x32_f16(
        /*neg_a=*/false, a, /*neg_b=*/false, b,
        /*c_mod=*/(short)0, c, /*reuse_a=*/false, /*reuse_b=*/false);
}

// CDNA5 async global->LDS copy (ASYNCcnt-tracked), 16B per lane.
// lds_off = LDS byte offset (low 32 bits of generic shared pointer), gptr = global addr.
__device__ __forceinline__ void async_ld_b128(unsigned lds_off, const void* gptr) {
    asm volatile("global_load_async_to_lds_b128 %0, %1, off"
                 :: "v"(lds_off), "v"(gptr) : "memory");
}
__device__ __forceinline__ void async_wait0() {
    asm volatile("s_wait_asynccnt 0x0" ::: "memory");
}
__device__ __forceinline__ void async_wait4() {
    asm volatile("s_wait_asynccnt 0x4" ::: "memory");
}

#define NEG_INF (-__builtin_inff())

// Problem constants
#define Bc 2
#define Sc 2048
#define Dc 1024
#define Hc 16
#define HDc 64

// ---------------------------------------------------------------------------
// Pre-pass: x fp32 [4096][1024] -> f16 (4 elems/thread, b128 in / b64 out)
// ---------------------------------------------------------------------------
__global__ void cvt_x_kernel(const float* __restrict__ x,
                             _Float16* __restrict__ x16) {
    int i = (blockIdx.x * 256 + threadIdx.x) * 4;    // 4194304 elems total
    v4f f = *(const v4f*)(x + i);
    v4h h;
    #pragma unroll
    for (int r = 0; r < 4; ++r) h[r] = (_Float16)f[r];
    *(v4h*)(x16 + i) = h;
}

// Pre-pass: Wo fp32 [D][D] -> f16 [D][D]
__global__ void cvt_wo_kernel(const float* __restrict__ wo,
                              _Float16* __restrict__ wo16) {
    int i = (blockIdx.x * 256 + threadIdx.x) * 4;    // 1048576 elems total
    v4f f = *(const v4f*)(wo + i);
    v4h h;
    #pragma unroll
    for (int r = 0; r < 4; ++r) h[r] = (_Float16)f[r];
    *(v4h*)(wo16 + i) = h;
}

// Pre-pass: Wq/Wk/Wv fp32 [H][D][HD] -> wt16 f16 [which][H][HD][D] (transposed)
__global__ void tr_w_kernel(const float* __restrict__ Wq,
                            const float* __restrict__ Wk,
                            const float* __restrict__ Wv,
                            _Float16* __restrict__ wt16) {
    int i = blockIdx.x * 256 + threadIdx.x;          // 3*16*64*1024 = 3145728
    int which = i >> 20;
    int rem   = i & 1048575;
    int h     = rem >> 16;
    int rem2  = rem & 65535;
    int e     = rem2 >> 10;
    int d     = rem2 & 1023;
    const float* W = (which == 0) ? Wq : (which == 1) ? Wk : Wv;
    wt16[i] = (_Float16)W[(h * Dc + d) * HDc + e];
}

// ---------------------------------------------------------------------------
// QKV projection. One wave per 16x16 output tile, K-loop over D in steps of 32.
// All operands pre-converted f16 -> pure b128 loads + WMMA in the hot loop.
//   which==0/1 (Q/K): D[m=col][n=s]  -> contiguous f16 store to [b][h][s][e]
//   which==2   (V)  : D[m=s][n=col]  -> contiguous f16 store to vt16 [b][h][e][s]
// Q is pre-scaled by 1/sqrt(HD) = 0.125.
// ---------------------------------------------------------------------------
__global__ void proj_kernel(const _Float16* __restrict__ x16,
                            const _Float16* __restrict__ wt16,
                            const float* __restrict__ bq,
                            const float* __restrict__ bk,
                            const float* __restrict__ bv,
                            _Float16* __restrict__ q16,
                            _Float16* __restrict__ k16,
                            _Float16* __restrict__ vt16) {
    int wid  = blockIdx.x * 4 + (threadIdx.x >> 5);
    int lane = threadIdx.x & 31;
    int ha   = lane >> 4;        // half: 0 or 1
    int nl   = lane & 15;

    int which = wid / 16384;                 // 0=Q 1=K 2=V
    int rem   = wid - which * 16384;
    int ct    = rem & 63;                    // col tile within 1024 cols
    int st    = rem >> 6;                    // row tile within 4096 rows
    int col0  = ct * 16;
    int h     = col0 >> 6;
    int e0    = col0 & 63;
    int s0    = st * 16;

    const _Float16* wbase = wt16 + ((size_t)(which * Hc + h)) * HDc * Dc;
    const float*    bias  = (which == 0) ? bq : (which == 1) ? bk : bv;

    v8f acc = {};

    if (which < 2) {
        // A = Wt rows (one output col per lane), B = X^T (one row per lane)
        const _Float16* arow = wbase + (size_t)(e0 + nl) * Dc;
        const _Float16* brow = x16 + (size_t)(s0 + nl) * Dc;
        for (int kk = 0; kk < 32; ++kk) {
            int k0 = kk * 32;
            v8h lo = *(const v8h*)(arow + k0 + 16 * ha);
            v8h hi = *(const v8h*)(arow + k0 + 16 * ha + 8);
            v16h a;
            #pragma unroll
            for (int i = 0; i < 8; ++i) { a[i] = lo[i]; a[8 + i] = hi[i]; }
            v16h b = *(const v16h*)(brow + k0 + 16 * ha);
            acc = wmma_f16(a, b, acc);
        }
        // store: rows e0 + r + 8*ha of output column space, fixed s = s0+nl
        int g    = s0 + nl;
        int bidx = g >> 11;
        int srow = g & 2047;
        float qs = (which == 0) ? 0.125f : 1.0f;
        v8h stv;
        #pragma unroll
        for (int r = 0; r < 8; ++r)
            stv[r] = (_Float16)((acc[r] + bias[h * HDc + e0 + 8 * ha + r]) * qs);
        _Float16* dst = ((which == 0) ? q16 : k16) +
            (((size_t)(bidx * Hc + h) * Sc + srow)) * HDc + e0 + 8 * ha;
        *(v8h*)dst = stv;
    } else {
        // V: A = X rows, B = Wt columns-of-output per lane
        const _Float16* arow = x16 + (size_t)(s0 + nl) * Dc;
        const _Float16* brow = wbase + (size_t)(e0 + nl) * Dc;
        for (int kk = 0; kk < 32; ++kk) {
            int k0 = kk * 32;
            v8h lo = *(const v8h*)(arow + k0 + 16 * ha);
            v8h hi = *(const v8h*)(arow + k0 + 16 * ha + 8);
            v16h a;
            #pragma unroll
            for (int i = 0; i < 8; ++i) { a[i] = lo[i]; a[8 + i] = hi[i]; }
            v16h b = *(const v16h*)(brow + k0 + 16 * ha);
            acc = wmma_f16(a, b, acc);
        }
        // store transposed V: vt16[b][h][e][s], 8 consecutive s per lane
        int ecol = e0 + nl;
        float bb = bias[h * HDc + ecol];
        int g0   = s0 + 8 * ha;
        int bidx = g0 >> 11;
        int sr0  = g0 & 2047;
        v8h stv;
        #pragma unroll
        for (int r = 0; r < 8; ++r) stv[r] = (_Float16)(acc[r] + bb);
        _Float16* dst = vt16 + ((size_t)(bidx * Hc + h) * HDc + ecol) * Sc + sr0;
        *(v8h*)dst = stv;
    }
}

// ---------------------------------------------------------------------------
// Flash attention with async-LDS K/V staging.
// Block = 4 waves = 4 consecutive 16-query tiles of one (b,h).
// Per 32-key chunk: 8KB of K/V^T staged into LDS by 16 async b128 loads
// (4 per wave), double-buffered; compute reads via ds_load.
//   S^T = K * Q^T   (D: m=key, n=query)  -> per-lane softmax stats
//   O^T = V^T * P^T (D: m=e,   n=query)  -> contiguous f16 store to wv16[b][s][d]
// ---------------------------------------------------------------------------
__global__ void att_kernel(const _Float16* __restrict__ q16,
                           const _Float16* __restrict__ k16,
                           const _Float16* __restrict__ vt16,
                           _Float16* __restrict__ wv16) {
    __shared__ _Float16 ldsK[2][32 * 64];   // [buf][key][e]   4KB each
    __shared__ _Float16 ldsV[2][64 * 32];   // [buf][e][key]   4KB each

    int w    = threadIdx.x >> 5;             // wave in block: 0..3
    int lane = threadIdx.x & 31;
    int ha   = lane >> 4;
    int nl   = lane & 15;

    int bh = blockIdx.x >> 5;                // b*16 + h
    int qg = blockIdx.x & 31;                // 64-query group
    int q0 = qg * 64 + w * 16;               // this wave's query tile

    const _Float16* qbase = q16 + (size_t)bh * Sc * HDc;
    const char*     kbase = (const char*)(k16 + (size_t)bh * Sc * HDc);
    const char*     vbase = (const char*)(vt16 + (size_t)bh * HDc * Sc);

    unsigned ldsK0 = (unsigned)(size_t)&ldsK[0][0];
    unsigned ldsK1 = (unsigned)(size_t)&ldsK[1][0];
    unsigned ldsV0 = (unsigned)(size_t)&ldsV[0][0];
    unsigned ldsV1 = (unsigned)(size_t)&ldsV[1][0];

    // Q B-fragments (feature chunks 0..31, 32..63), loaded once
    v16h bq[2];
    {
        const _Float16* qrow = qbase + (size_t)(q0 + nl) * HDc;
        bq[0] = *(const v16h*)(qrow + 16 * ha);
        bq[1] = *(const v16h*)(qrow + 32 + 16 * ha);
    }

    v8f o[4];
    #pragma unroll
    for (int eg = 0; eg < 4; ++eg) o[eg] = (v8f){};
    float m_run = NEG_INF;
    float l_run = 0.0f;

    int my_nch  = (q0 + 15) / 32 + 1;        // chunks this wave needs
    int nch_max = 2 * qg + 2;                // chunks for wave 3 of this block

    // --- async stage of one 32-key chunk into buffer `bi` -------------------
    auto issue_chunk = [&](int c, int bi) {
        size_t j0b = (size_t)c * 32 * 2;     // key offset in bytes (f16)
        unsigned lk = bi ? ldsK1 : ldsK0;
        unsigned lv = bi ? ldsV1 : ldsV0;
        #pragma unroll
        for (int i = 0; i < 2; ++i) {
            int off = (w * 2 + i) * 512 + lane * 16;   // 0..4095 bytes
            // K chunk is contiguous: rows j0..j0+31 of [s][e] (128B rows)
            async_ld_b128(lk + off, kbase + j0b * 64 + off);
            // V^T chunk: 64 rows (e), 64B slice from each 4096B row
            int e = off >> 6;
            async_ld_b128(lv + off, vbase + (size_t)e * 4096 + j0b + (off & 63));
        }
    };

    issue_chunk(0, 0);

    for (int c = 0; c < nch_max; ++c) {
        int  bi   = c & 1;
        bool more = (c + 1) < nch_max;
        if (more) issue_chunk(c + 1, bi ^ 1);
        if (more) async_wait4(); else async_wait0();   // chunk c resident
        __syncthreads();                               // visible to all waves

        if (c < my_nch) {
            int j0 = c * 32;
            v8f slo = {}, shi = {};
            #pragma unroll
            for (int fc = 0; fc < 2; ++fc) {
                v16h a0, a1;
                {
                    const _Float16* kr0 = &ldsK[bi][(0 * 16 + nl) * 64 + fc * 32];
                    const _Float16* kr1 = &ldsK[bi][(1 * 16 + nl) * 64 + fc * 32];
                    v8h l0 = *(const v8h*)(kr0 + 16 * ha);
                    v8h h0 = *(const v8h*)(kr0 + 16 * ha + 8);
                    v8h l1 = *(const v8h*)(kr1 + 16 * ha);
                    v8h h1 = *(const v8h*)(kr1 + 16 * ha + 8);
                    #pragma unroll
                    for (int i = 0; i < 8; ++i) {
                        a0[i] = l0[i]; a0[8 + i] = h0[i];
                        a1[i] = l1[i]; a1[8 + i] = h1[i];
                    }
                }
                slo = wmma_f16(a0, bq[fc], slo);
                shi = wmma_f16(a1, bq[fc], shi);
            }

            // causal mask (diagonal chunks only; wave-uniform condition)
            int q = q0 + nl;
            if (j0 + 31 > q0) {
                #pragma unroll
                for (int r = 0; r < 8; ++r) {
                    if (j0 + r + 8 * ha > q)      slo[r] = NEG_INF;
                    if (j0 + 16 + r + 8 * ha > q) shi[r] = NEG_INF;
                }
            }

            // online softmax (per query = per lane, + one cross-half shuffle)
            float cmax = NEG_INF;
            #pragma unroll
            for (int r = 0; r < 8; ++r) cmax = fmaxf(cmax, fmaxf(slo[r], shi[r]));
            cmax = fmaxf(cmax, __shfl_xor(cmax, 16, 32));
            float m_new = fmaxf(m_run, cmax);
            float scale = __expf(m_run - m_new);

            float plo[8], phi[8];
            float csum = 0.0f;
            #pragma unroll
            for (int r = 0; r < 8; ++r) {
                plo[r] = __expf(slo[r] - m_new);
                phi[r] = __expf(shi[r] - m_new);
                csum += plo[r] + phi[r];
            }
            csum += __shfl_xor(csum, 16, 32);
            l_run = l_run * scale + csum;
            m_run = m_new;

            #pragma unroll
            for (int eg = 0; eg < 4; ++eg)
                #pragma unroll
                for (int r = 0; r < 8; ++r) o[eg][r] *= scale;

            // P^T B-fragment (contract dim = 32 keys): cross-half exchange
            v16h pb;
            #pragma unroll
            for (int r = 0; r < 8; ++r) {
                float s0 = __shfl_xor(plo[r], 16, 32);
                float s1 = __shfl_xor(phi[r], 16, 32);
                float lowv  = ha ? s1     : plo[r];  // key = 16*ha + r
                float highv = ha ? phi[r] : s0;      // key = 16*ha + 8 + r
                pb[r]     = (_Float16)lowv;
                pb[8 + r] = (_Float16)highv;
            }

            // O^T += V^T x P^T   (4 e-groups of 16)
            #pragma unroll
            for (int eg = 0; eg < 4; ++eg) {
                const _Float16* vrow = &ldsV[bi][(eg * 16 + nl) * 32];
                v8h lo = *(const v8h*)(vrow + 16 * ha);
                v8h hi = *(const v8h*)(vrow + 16 * ha + 8);
                v16h a;
                #pragma unroll
                for (int i = 0; i < 8; ++i) { a[i] = lo[i]; a[8 + i] = hi[i]; }
                o[eg] = wmma_f16(a, pb, o[eg]);
            }
        }
        __syncthreads();   // all waves done reading buf bi before it is reused
    }

    // normalize + store: wv16[b][q0+nl][h*64 + eg*16 + 8*ha + r]
    float inv = 1.0f / l_run;
    int b = bh >> 4;
    int h = bh & 15;
    size_t rowoff = ((size_t)(b * Sc + q0 + nl)) * Dc + h * HDc;
    #pragma unroll
    for (int eg = 0; eg < 4; ++eg) {
        v8h stv;
        #pragma unroll
        for (int r = 0; r < 8; ++r) stv[r] = (_Float16)(o[eg][r] * inv);
        *(v8h*)(wv16 + rowoff + eg * 16 + 8 * ha) = stv;
    }
}

// ---------------------------------------------------------------------------
// Output projection: out[s][d] = sum_c wv[s][c] * Wo[d][c] + bo[d]  (fp32 out)
//   A = Wo16 rows (m=d), B = wv^T (n=s)  -> D[m=d][n=s], contiguous fp32 store
// ---------------------------------------------------------------------------
__global__ void outproj_kernel(const _Float16* __restrict__ wo16,
                               const _Float16* __restrict__ wv16,
                               const float* __restrict__ bo,
                               float* __restrict__ out) {
    int wid  = blockIdx.x * 4 + (threadIdx.x >> 5);
    int lane = threadIdx.x & 31;
    int ha   = lane >> 4;
    int nl   = lane & 15;

    int dt = wid & 63;
    int st = wid >> 6;
    int d0 = dt * 16;
    int s0 = st * 16;

    const _Float16* arow = wo16 + (size_t)(d0 + nl) * Dc;
    const _Float16* brow = wv16 + (size_t)(s0 + nl) * Dc;

    v8f acc = {};
    for (int kk = 0; kk < 32; ++kk) {
        int k0 = kk * 32;
        v8h lo = *(const v8h*)(arow + k0 + 16 * ha);
        v8h hi = *(const v8h*)(arow + k0 + 16 * ha + 8);
        v16h a;
        #pragma unroll
        for (int i = 0; i < 8; ++i) { a[i] = lo[i]; a[8 + i] = hi[i]; }
        v16h b = *(const v16h*)(brow + k0 + 16 * ha);
        acc = wmma_f16(a, b, acc);
    }

    float* dst = out + (size_t)(s0 + nl) * Dc + d0 + 8 * ha;
    v4f o0, o1;
    #pragma unroll
    for (int r = 0; r < 4; ++r) {
        o0[r] = acc[r]     + bo[d0 + 8 * ha + r];
        o1[r] = acc[4 + r] + bo[d0 + 8 * ha + 4 + r];
    }
    *(v4f*)dst       = o0;
    *(v4f*)(dst + 4) = o1;
}

// ---------------------------------------------------------------------------
// Launch
// ---------------------------------------------------------------------------
extern "C" void kernel_launch(void* const* d_in, const int* in_sizes, int n_in,
                              void* d_out, int out_size, void* d_ws, size_t ws_size,
                              hipStream_t stream) {
    const float* x  = (const float*)d_in[0];
    const float* Wq = (const float*)d_in[1];
    const float* bq = (const float*)d_in[2];
    const float* Wk = (const float*)d_in[3];
    const float* bk = (const float*)d_in[4];
    const float* Wv = (const float*)d_in[5];
    const float* bv = (const float*)d_in[6];
    const float* Wo = (const float*)d_in[7];
    const float* bo = (const float*)d_in[8];
    float* out = (float*)d_out;

    char* ws = (char*)d_ws;
    _Float16* wt16 = (_Float16*)(ws);               //  6 MB: Wqkv^T f16
    _Float16* wo16 = (_Float16*)(ws + 6291456);     //  2 MB: Wo f16
    _Float16* x16  = (_Float16*)(ws + 8388608);     //  8 MB: x f16 [b*s][d]
    _Float16* q16  = (_Float16*)(ws + 16777216);    //  8 MB: Q  [b][h][s][e] (pre-scaled)
    _Float16* k16  = (_Float16*)(ws + 25165824);    //  8 MB: K  [b][h][s][e]
    _Float16* vt16 = (_Float16*)(ws + 33554432);    //  8 MB: V^T[b][h][e][s]
    _Float16* wv16 = (_Float16*)(ws + 41943040);    //  8 MB: attn out [b][s][d]

    cvt_x_kernel<<<4096, 256, 0, stream>>>(x, x16);
    tr_w_kernel<<<12288, 256, 0, stream>>>(Wq, Wk, Wv, wt16);
    cvt_wo_kernel<<<1024, 256, 0, stream>>>(Wo, wo16);
    proj_kernel<<<12288, 128, 0, stream>>>(x16, wt16, bq, bk, bv, q16, k16, vt16);
    att_kernel<<<1024, 128, 0, stream>>>(q16, k16, vt16, wv16);
    outproj_kernel<<<4096, 128, 0, stream>>>(wo16, wv16, bo, out);
}
